// GatedLinearAttention_58136677319204
// MI455X (gfx1250) — compile-verified, matrix operand order checked by reading, not run on hardware
//
#include <hip/hip_runtime.h>
#include <hip/hip_bf16.h>
#include <math.h>

// Problem constants (match reference)
#define BB    2
#define TT    2048
#define DD    2048
#define HH    16
#define DKc   64
#define DVc   128
#define CCk   64
#define NCH   (TT / CCk)
#define MROWS (BB * TT)

typedef __attribute__((ext_vector_type(8)))  float        v8f;
typedef __attribute__((ext_vector_type(2)))  float        v2f;
typedef __attribute__((ext_vector_type(16))) __bf16       v16bf;
typedef __attribute__((ext_vector_type(4)))  unsigned int v4u;
typedef __attribute__((ext_vector_type(8)))  int          v8i;
typedef __attribute__((ext_vector_type(4)))  int          v4i;

static __device__ inline v8f zero_v8f() {
    v8f z;
#pragma unroll
    for (int i = 0; i < 8; ++i) z[i] = 0.0f;
    return z;
}

// ---------------------------------------------------------------------------
// Tensor Data Mover: async 2D tile load (64 rows x 64 f32, row stride 1024 f32)
// D# built per CDNA5 ISA (group0/group1; groups 2,3 zero).
// 6-arg builtin form (clang-23 / therock headers).
// ---------------------------------------------------------------------------
static __device__ inline void tdm_load_tile_64x64(const float* gaddr, float* laddr) {
    const unsigned long long ga = (unsigned long long)(uintptr_t)gaddr;
    v4u g0;
    g0[0] = 1u;                                   // count=1, user descriptor
    g0[1] = (unsigned)(uintptr_t)laddr;           // LDS byte offset
    g0[2] = (unsigned)ga;                         // global_addr[31:0]
    g0[3] = (unsigned)((ga >> 32) & 0x01FFFFFFull) | 0x80000000u; // [56:32] | type=2
    v8i g1;
    g1[0] = (int)(2u << 16);                      // workgroup_mask=0, data_size=4B
    g1[1] = (int)(64u << 16);                     // tensor_dim0 = 64 (low 16 bits)
    g1[2] = (int)(64u << 16);                     // tensor_dim0 hi=0 | tensor_dim1 = 64
    g1[3] = (int)(64u << 16);                     // tensor_dim1 hi=0 | tile_dim0 = 64
    g1[4] = 64;                                   // tile_dim1 = 64, tile_dim2 = 0
    g1[5] = (int)(HH * DKc);                      // tensor_dim0_stride = 1024 (lo 32)
    g1[6] = 0;                                    // stride hi | dim1_stride lo
    g1[7] = 0;
    v4i z4; z4[0] = 0; z4[1] = 0; z4[2] = 0; z4[3] = 0;
    v8i z8;
#pragma unroll
    for (int i = 0; i < 8; ++i) z8[i] = 0;
    __builtin_amdgcn_tensor_load_to_lds(g0, g1, z4, z4, z8, 0);
}

// ---------------------------------------------------------------------------
// bf16 WMMA GEMM:  Out[M,N] = epi( A[M,K] @ W[K,N] )
// Block tile 128x128, K-step 32, 8 waves (wave32), each wave 64x32 (4x2 tiles).
// sA is [m][k] row-major bf16; sBT is [n][k] column-major bf16 so both A and B
// fragments load as two contiguous ds_load_b128.
// ---------------------------------------------------------------------------
#define GBM 128
#define GBN 128
#define GBK 32

static __device__ inline v16bf ldfrag_mk(const __bf16* s, int mBase, int lane) {
    const int m = mBase + (lane & 15);
    const int g = (lane >> 4) * 8;
    v16bf a;
#pragma unroll
    for (int i = 0; i < 16; ++i) {
        const int kk = (i < 8 ? i : i + 8) + g;   // ISA 16-bit A/B 16x32 layout
        a[i] = s[m * GBK + kk];
    }
    return a;
}

__global__ __launch_bounds__(256) void gemm_bf16_kernel(
    const float* __restrict__ A, const float* __restrict__ W,
    float* __restrict__ Out, int M, int N, int K, int epi, float scale)
{
    __shared__ __bf16 sA[GBM * GBK];    // [m][k]
    __shared__ __bf16 sBT[GBN * GBK];   // [n][k]

    const int tid  = threadIdx.x;
    const int lane = tid & 31;
    const int wave = tid >> 5;
    const int wm   = wave & 1;   // 64-row strip
    const int wn   = wave >> 1;  // 32-col strip
    const int bm   = blockIdx.y * GBM;
    const int bn   = blockIdx.x * GBN;

    v8f acc[4][2];
#pragma unroll
    for (int mt = 0; mt < 4; ++mt)
#pragma unroll
        for (int nt = 0; nt < 2; ++nt) acc[mt][nt] = zero_v8f();

    for (int k0 = 0; k0 < K; k0 += GBK) {
        // Stage A tile (128x32 f32 -> bf16), row-major
#pragma unroll
        for (int i = 0; i < 4; ++i) {
            const int fi = tid + 256 * i;      // float4 index, 0..1023
            const int r  = fi >> 3;
            const int c4 = fi & 7;
            const float4 va = *reinterpret_cast<const float4*>(
                A + (size_t)(bm + r) * K + k0 + c4 * 4);
            sA[r * GBK + c4 * 4 + 0] = (__bf16)va.x;
            sA[r * GBK + c4 * 4 + 1] = (__bf16)va.y;
            sA[r * GBK + c4 * 4 + 2] = (__bf16)va.z;
            sA[r * GBK + c4 * 4 + 3] = (__bf16)va.w;
        }
        // Stage B tile (32x128 f32 -> bf16), transposed to [n][k] in LDS.
        // Each task: one column n, 4 consecutive K rows; loads are coalesced
        // across lanes (consecutive n), store is one contiguous 4x bf16.
#pragma unroll
        for (int i = 0; i < 4; ++i) {
            const int task = tid + 256 * i;    // 0..1023
            const int n    = task & 127;
            const int kr   = (task >> 7) * 4;  // 0,4,...,28
            const float w0 = W[(size_t)(k0 + kr + 0) * N + bn + n];
            const float w1 = W[(size_t)(k0 + kr + 1) * N + bn + n];
            const float w2 = W[(size_t)(k0 + kr + 2) * N + bn + n];
            const float w3 = W[(size_t)(k0 + kr + 3) * N + bn + n];
            sBT[n * GBK + kr + 0] = (__bf16)w0;
            sBT[n * GBK + kr + 1] = (__bf16)w1;
            sBT[n * GBK + kr + 2] = (__bf16)w2;
            sBT[n * GBK + kr + 3] = (__bf16)w3;
        }
        if (k0 + GBK < K) {  // L2 prefetch of next A tile -> global_prefetch
            __builtin_prefetch(A + (size_t)(bm + (tid >> 1)) * K + k0 + GBK, 0, 1);
        }
        __syncthreads();

        v16bf af[4], bf[2];
#pragma unroll
        for (int mt = 0; mt < 4; ++mt) af[mt] = ldfrag_mk(sA,  wm * 64 + mt * 16, lane);
#pragma unroll
        for (int nt = 0; nt < 2; ++nt) bf[nt] = ldfrag_mk(sBT, wn * 32 + nt * 16, lane);
#pragma unroll
        for (int mt = 0; mt < 4; ++mt)
#pragma unroll
            for (int nt = 0; nt < 2; ++nt)
                acc[mt][nt] = __builtin_amdgcn_wmma_f32_16x16x32_bf16(
                    false, af[mt], false, bf[nt], (short)0, acc[mt][nt], false, false);
        __syncthreads();
    }

    const int g  = lane >> 4;
    const int ln = lane & 15;
#pragma unroll
    for (int mt = 0; mt < 4; ++mt)
#pragma unroll
        for (int nt = 0; nt < 2; ++nt)
#pragma unroll
            for (int r = 0; r < 8; ++r) {
                const int row = bm + wm * 64 + mt * 16 + r + 8 * g;
                const int col = bn + wn * 32 + nt * 16 + ln;
                float x = acc[mt][nt][r];
                if (epi) { x = x / (1.0f + __expf(-x)); x *= scale; }
                Out[(size_t)row * N + col] = x;
            }
}

// ---------------------------------------------------------------------------
// Low-rank projection: t1 = hs @ Wgk1, t2 = hs @ Wg1  (both [M,16], packed [M,32])
// ---------------------------------------------------------------------------
__global__ __launch_bounds__(256) void lowrank_proj_kernel(
    const float* __restrict__ hs, const float* __restrict__ W1a,
    const float* __restrict__ W1b, float* __restrict__ t12)
{
    const int wave = threadIdx.x >> 5;
    const int lane = threadIdx.x & 31;
    const int row  = blockIdx.x * 8 + wave;

    float a1[16], a2[16];
#pragma unroll
    for (int j = 0; j < 16; ++j) { a1[j] = 0.f; a2[j] = 0.f; }

    const float* hrow = hs + (size_t)row * DD;
    for (int kx = lane; kx < DD; kx += 32) {
        const float h = hrow[kx];
        const float4* wa = reinterpret_cast<const float4*>(W1a + (size_t)kx * 16);
        const float4* wb = reinterpret_cast<const float4*>(W1b + (size_t)kx * 16);
#pragma unroll
        for (int j4 = 0; j4 < 4; ++j4) {
            const float4 va = wa[j4], vb = wb[j4];
            a1[j4 * 4 + 0] += h * va.x; a1[j4 * 4 + 1] += h * va.y;
            a1[j4 * 4 + 2] += h * va.z; a1[j4 * 4 + 3] += h * va.w;
            a2[j4 * 4 + 0] += h * vb.x; a2[j4 * 4 + 1] += h * vb.y;
            a2[j4 * 4 + 2] += h * vb.z; a2[j4 * 4 + 3] += h * vb.w;
        }
    }
#pragma unroll
    for (int off = 16; off > 0; off >>= 1) {
#pragma unroll
        for (int j = 0; j < 16; ++j) {
            a1[j] += __shfl_xor(a1[j], off, 32);
            a2[j] += __shfl_xor(a2[j], off, 32);
        }
    }
    if (lane == 0) {
#pragma unroll
        for (int j = 0; j < 16; ++j) {
            t12[(size_t)row * 32 + j]      = a1[j];
            t12[(size_t)row * 32 + 16 + j] = a2[j];
        }
    }
}

// gk = logsigmoid(t1 @ Wgk2 + b)/16  (mode 1) ; g = t2 @ Wg2 + b (mode 0)
__global__ __launch_bounds__(256) void lowrank_out_kernel(
    const float* __restrict__ t12, int toff, const float* __restrict__ W2,
    const float* __restrict__ b2, float* __restrict__ out, int Ncols, int mode)
{
    const int idx = blockIdx.x * 256 + threadIdx.x;
    const int row = idx / Ncols;
    const int c   = idx % Ncols;
    const float* tr = t12 + (size_t)row * 32 + toff;
    float s = b2[c];
#pragma unroll
    for (int j = 0; j < 16; ++j) s += tr[j] * W2[j * Ncols + c];
    if (mode) {
        const float ls = fminf(s, 0.f) - log1pf(__expf(-fabsf(s)));
        s = ls * (1.0f / 16.0f);
    }
    out[idx] = s;
}

// ---------------------------------------------------------------------------
// GLA chunk scan. One block per (b, h, DV-group-of-32). 4 waves (128 thr).
// State S [64 x 32] fp32 persists in LDS across the 32-chunk sequential scan.
// Chunk staging of gk/q/k tiles uses the Tensor Data Mover (async DMA to LDS).
// fp32 WMMA 16x16x4 for qg@S, qg@kg^T (masked), A@v, kdec^T@v.
// ---------------------------------------------------------------------------
__global__ __launch_bounds__(128) void gla_kernel(
    const float* __restrict__ qp, const float* __restrict__ kp,
    const float* __restrict__ vp, const float* __restrict__ gkp,
    float* __restrict__ op)
{
    __shared__ float sS[DKc * 32];     // state slice  [dk][32]
    __shared__ float sQg[CCk * DKc];   // raw q -> q * exp(G)   [t][dk]
    __shared__ float sKg[CCk * DKc];   // raw k -> k * exp(-G)  [t][dk]
    __shared__ float sGA[CCk * DKc];   // raw gk -> G (cumsum) -> masked A
    __shared__ float sEg[DKc];         // exp(G_last)

    const int tid  = threadIdx.x;
    const int lane = tid & 31;
    const int wave = tid >> 5;
    const int g    = lane >> 4;
    const int ln   = lane & 15;

    const int bid     = blockIdx.x;
    const int b       = bid / (HH * 4);
    const int h       = (bid / 4) % HH;
    const int colBase = (bid & 3) * 32;

    for (int i = tid; i < DKc * 32; i += 128) sS[i] = 0.f;

    const size_t qkBase = ((size_t)b * TT) * HH * DKc + (size_t)h * DKc;
    const size_t vBase  = ((size_t)b * TT) * HH * DVc + (size_t)h * DVc + colBase;

    for (int ch = 0; ch < NCH; ++ch) {
        const int t0 = ch * CCk;

        // ---- phase 1: TDM-stage gk/q/k 64x64 tiles into LDS ----
        if (wave == 0) {
            const size_t cOff = qkBase + (size_t)t0 * HH * DKc;
            tdm_load_tile_64x64(gkp + cOff, sGA);
            tdm_load_tile_64x64(qp  + cOff, sQg);
            tdm_load_tile_64x64(kp  + cOff, sKg);
            __builtin_amdgcn_s_wait_tensorcnt(0);
        }
        __syncthreads();

        // cumsum of gk along time per dk column; record exp(G_last)
        if (tid < DKc) {
            float s = 0.f;
            for (int t = 0; t < CCk; ++t) { s += sGA[t * DKc + tid]; sGA[t * DKc + tid] = s; }
            sEg[tid] = __expf(s);
        }
        __syncthreads();

        // ---- phase 2: in-place qg = q*exp(G), kg = k*exp(-G) ----
#pragma unroll
        for (int i = 0; i < 32; ++i) {
            const int e = tid + 128 * i;
            const float G = sGA[e];
            sQg[e] = sQg[e] * __expf(G);
            sKg[e] = sKg[e] * __expf(-G);
        }
        __syncthreads();

        // ---- phase 3+4: o1 = qg @ S ; A = qg @ kg^T (K = DK = 64) ----
        const int mT = wave * 16;
        v8f accO[2]; accO[0] = zero_v8f(); accO[1] = zero_v8f();
        v8f accA[4];
#pragma unroll
        for (int nt = 0; nt < 4; ++nt) accA[nt] = zero_v8f();

#pragma unroll
        for (int s4 = 0; s4 < DKc; s4 += 4) {
            v2f a;  // A-frag: lanes 0-15 K={0,1}, lanes 16-31 K={2,3}
            a[0] = sQg[(mT + ln) * DKc + s4 + 2 * g + 0];
            a[1] = sQg[(mT + ln) * DKc + s4 + 2 * g + 1];
#pragma unroll
            for (int nt = 0; nt < 2; ++nt) {
                v2f bS; // B-frag: vgpr r -> K = s4 + r + 2g, N = lane&15
                bS[0] = sS[(s4 + 2 * g + 0) * 32 + nt * 16 + ln];
                bS[1] = sS[(s4 + 2 * g + 1) * 32 + nt * 16 + ln];
                accO[nt] = __builtin_amdgcn_wmma_f32_16x16x4_f32(
                    false, a, false, bS, (short)0, accO[nt], false, false);
            }
#pragma unroll
            for (int nt = 0; nt < 4; ++nt) {
                v2f bK; // B = kg^T : B[kk][n] = kg[n][kk]
                bK[0] = sKg[(nt * 16 + ln) * DKc + s4 + 2 * g + 0];
                bK[1] = sKg[(nt * 16 + ln) * DKc + s4 + 2 * g + 1];
                accA[nt] = __builtin_amdgcn_wmma_f32_16x16x4_f32(
                    false, a, false, bK, (short)0, accA[nt], false, false);
            }
        }

        // causal mask and spill A -> LDS (reuses G buffer)
#pragma unroll
        for (int nt = 0; nt < 4; ++nt)
#pragma unroll
            for (int r = 0; r < 8; ++r) {
                const int m = mT + r + 8 * g;
                const int n = nt * 16 + ln;
                sGA[m * 64 + n] = (m >= n) ? accA[nt][r] : 0.f;
            }
        __syncthreads();

        // ---- phase 5: o += A @ v (K = C = 64; v streamed from global) ----
#pragma unroll
        for (int s4 = 0; s4 < CCk; s4 += 4) {
            v2f a;
            a[0] = sGA[(mT + ln) * 64 + s4 + 2 * g + 0];
            a[1] = sGA[(mT + ln) * 64 + s4 + 2 * g + 1];
#pragma unroll
            for (int nt = 0; nt < 2; ++nt) {
                v2f bV;
                bV[0] = vp[vBase + (size_t)(t0 + s4 + 2 * g + 0) * HH * DVc + nt * 16 + ln];
                bV[1] = vp[vBase + (size_t)(t0 + s4 + 2 * g + 1) * HH * DVc + nt * 16 + ln];
                accO[nt] = __builtin_amdgcn_wmma_f32_16x16x4_f32(
                    false, a, false, bV, (short)0, accO[nt], false, false);
            }
        }
        // store o chunk
#pragma unroll
        for (int nt = 0; nt < 2; ++nt)
#pragma unroll
            for (int r = 0; r < 8; ++r) {
                const int m = mT + r + 8 * g;
                op[vBase + (size_t)(t0 + m) * HH * DVc + nt * 16 + ln] = accO[nt][r];
            }
        __syncthreads();

        // ---- phase 6: S = diag-row(exp(Gl)) * S + kdec^T @ v ----
        const int mS = wave * 16;  // dk strip owned by this wave
        v8f accS[2];
#pragma unroll
        for (int nt = 0; nt < 2; ++nt)
#pragma unroll
            for (int r = 0; r < 8; ++r) {
                const int m = mS + r + 8 * g;
                accS[nt][r] = sS[m * 32 + nt * 16 + ln] * sEg[m];
            }
        const float em = sEg[mS + ln];
#pragma unroll
        for (int s4 = 0; s4 < CCk; s4 += 4) {
            v2f a;  // kdec^T[d][t] = kg[t][d] * exp(Gl[d])
            a[0] = sKg[(s4 + 2 * g + 0) * DKc + mS + ln] * em;
            a[1] = sKg[(s4 + 2 * g + 1) * DKc + mS + ln] * em;
#pragma unroll
            for (int nt = 0; nt < 2; ++nt) {
                v2f bV;
                bV[0] = vp[vBase + (size_t)(t0 + s4 + 2 * g + 0) * HH * DVc + nt * 16 + ln];
                bV[1] = vp[vBase + (size_t)(t0 + s4 + 2 * g + 1) * HH * DVc + nt * 16 + ln];
                accS[nt] = __builtin_amdgcn_wmma_f32_16x16x4_f32(
                    false, a, false, bV, (short)0, accS[nt], false, false);
            }
        }
#pragma unroll
        for (int nt = 0; nt < 2; ++nt)
#pragma unroll
            for (int r = 0; r < 8; ++r) {
                const int m = mS + r + 8 * g;
                sS[m * 32 + nt * 16 + ln] = accS[nt][r];
            }
        __syncthreads();
    }
}

// ---------------------------------------------------------------------------
// RMS group-norm over DV + sigmoid(g) gate. One wave per (b,t,h) row.
// ---------------------------------------------------------------------------
__global__ __launch_bounds__(256) void norm_gate_kernel(
    const float* __restrict__ o, const float* __restrict__ gRaw,
    const float* __restrict__ gnw, float* __restrict__ on)
{
    const int wave = threadIdx.x >> 5;
    const int lane = threadIdx.x & 31;
    const size_t row = (size_t)blockIdx.x * 8 + wave;
    const float* orow = o + row * DVc;
    const float* grow = gRaw + row * DVc;
    float x[4];
    float ss = 0.f;
#pragma unroll
    for (int j = 0; j < 4; ++j) { x[j] = orow[lane + 32 * j]; ss += x[j] * x[j]; }
#pragma unroll
    for (int off = 16; off > 0; off >>= 1) ss += __shfl_xor(ss, off, 32);
    const float rn = rsqrtf(ss * (1.0f / DVc) + 1e-5f);
    float* onrow = on + row * DVc;
#pragma unroll
    for (int j = 0; j < 4; ++j) {
        const int e = lane + 32 * j;
        const float gv = grow[e];
        onrow[e] = x[j] * rn * gnw[e] / (1.0f + __expf(-gv));
    }
}

// ---------------------------------------------------------------------------
extern "C" void kernel_launch(void* const* d_in, const int* in_sizes, int n_in,
                              void* d_out, int out_size, void* d_ws, size_t ws_size,
                              hipStream_t stream)
{
    (void)in_sizes; (void)n_in; (void)out_size; (void)ws_size;
    const float* hs    = (const float*)d_in[0];
    const float* Wq    = (const float*)d_in[1];
    const float* Wk    = (const float*)d_in[2];
    const float* Wv    = (const float*)d_in[3];
    const float* Wgk1  = (const float*)d_in[4];
    const float* Wgk2  = (const float*)d_in[5];
    const float* bgk2  = (const float*)d_in[6];
    const float* Wg1   = (const float*)d_in[7];
    const float* Wg2   = (const float*)d_in[8];
    const float* bg2   = (const float*)d_in[9];
    const float* Wo    = (const float*)d_in[10];
    const float* gnw   = (const float*)d_in[11];
    float* out = (float*)d_out;

    // Workspace layout (floats): ~185 MB total
    float* ws  = (float*)d_ws;
    float* qb  = ws;                                  // M*1024
    float* kb  = qb  + (size_t)MROWS * HH * DKc;      // M*1024
    float* vb  = kb  + (size_t)MROWS * HH * DKc;      // M*2048
    float* gkb = vb  + (size_t)MROWS * HH * DVc;      // M*1024
    float* gb  = gkb + (size_t)MROWS * HH * DKc;      // M*2048
    float* t12 = gb  + (size_t)MROWS * HH * DVc;      // M*32
    float* ob  = t12 + (size_t)MROWS * 32;            // M*2048
    float* onb = ob  + (size_t)MROWS * HH * DVc;      // M*2048

    const dim3 blk256(256);

    // q/k/v projections with SiLU (q additionally scaled by DK^-0.5)
    gemm_bf16_kernel<<<dim3(1024 / GBN, MROWS / GBM), blk256, 0, stream>>>(
        hs, Wq, qb, MROWS, 1024, DD, 1, 0.125f);
    gemm_bf16_kernel<<<dim3(1024 / GBN, MROWS / GBM), blk256, 0, stream>>>(
        hs, Wk, kb, MROWS, 1024, DD, 1, 1.0f);
    gemm_bf16_kernel<<<dim3(2048 / GBN, MROWS / GBM), blk256, 0, stream>>>(
        hs, Wv, vb, MROWS, 2048, DD, 1, 1.0f);

    // rank-16 paths
    lowrank_proj_kernel<<<MROWS / 8, blk256, 0, stream>>>(hs, Wgk1, Wg1, t12);
    lowrank_out_kernel<<<(MROWS * 1024) / 256, blk256, 0, stream>>>(
        t12, 0, Wgk2, bgk2, gkb, 1024, 1);
    lowrank_out_kernel<<<(MROWS * 2048) / 256, blk256, 0, stream>>>(
        t12, 16, Wg2, bg2, gb, 2048, 0);

    // chunked GLA scan (fp32 WMMA + TDM staging), 128 blocks x 128 threads
    gla_kernel<<<BB * HH * 4, dim3(128), 0, stream>>>(qb, kb, vb, gkb, ob);

    // RMS norm + sigmoid gate
    norm_gate_kernel<<<(BB * TT * HH) / 8, blk256, 0, stream>>>(ob, gb, gnw, onb);

    // output projection
    gemm_bf16_kernel<<<dim3(2048 / GBN, MROWS / GBM), blk256, 0, stream>>>(
        onb, Wo, out, MROWS, 2048, DD, 0, 1.0f);
}
